// MultiHeadSelfAttention_7215545057789
// MI455X (gfx1250) — matrix operand order repaired and checked by reading, not compiled, not run on hardware
//
#include <hip/hip_runtime.h>
#include <hip/hip_bf16.h>

typedef __bf16 bf16_t;
typedef __attribute__((ext_vector_type(16))) __bf16 v16bf;
typedef __attribute__((ext_vector_type(8)))  __bf16 v8bf;
typedef __attribute__((ext_vector_type(8)))  float  v8f;
typedef __attribute__((ext_vector_type(4)))  float  v4f;

#define D_MODEL 1024
#define SEQ     2048
#define BATCH   2
#define NH      16
#define DK      64
#define NTOK    (BATCH*SEQ)   /* 4096 */

// ---------------- WMMA helpers ----------------

static __device__ inline v8f wmma_bf16(v16bf a, v16bf b, v8f c) {
  // D = A(16x32 bf16) * B(32x16 bf16) + C(16x16 f32)
  return __builtin_amdgcn_wmma_f32_16x16x32_bf16(false, a, false, b, (short)0, c,
                                                 false, false);
}

// A-fragment (16x32 bf16): lane holds row m = lane%16.
// lane<16 : K = {0..15}; lane>=16 : K = {16..31}
// p points at row start + K-chunk base (already includes m*ld + kk).
static __device__ inline v16bf load_a_frag_bf16(const bf16_t* __restrict__ p, int hi) {
  const bf16_t* q = p + (hi ? 16 : 0);
  v8bf g0 = *(const v8bf*)(q);
  v8bf g1 = *(const v8bf*)(q + 8);
  v16bf a;
#pragma unroll
  for (int i = 0; i < 8; ++i) {
    a[i]     = g0[i];
    a[8 + i] = g1[i];
  }
  return a;
}

// B-fragment (32x16): lane holds column n = lane%16; elements j -> K = j + (hi?16:0).
// B = W^T with W row-major => 16 contiguous bf16 of W row n (one 32B load).
static __device__ inline v16bf load_b_frag_bf16(const bf16_t* __restrict__ p, int hi) {
  return *(const v16bf*)(p + (hi ? 16 : 0));
}

// 16-lane (half-wave) reductions: masks <16 never cross the 16-lane groups.
static __device__ inline float halfmax16(float v) {
  v = fmaxf(v, __shfl_xor(v, 1, 32));
  v = fmaxf(v, __shfl_xor(v, 2, 32));
  v = fmaxf(v, __shfl_xor(v, 4, 32));
  v = fmaxf(v, __shfl_xor(v, 8, 32));
  return v;
}
static __device__ inline float halfsum16(float v) {
  v += __shfl_xor(v, 1, 32);
  v += __shfl_xor(v, 2, 32);
  v += __shfl_xor(v, 4, 32);
  v += __shfl_xor(v, 8, 32);
  return v;
}

// ---------------- Kernel 0: one-time f32 -> bf16 conversion ----------------
// 8 elements / thread; element count must be a multiple of 2048 (it is).
__global__ __launch_bounds__(256) void convert_f32_bf16_kernel(
    const float* __restrict__ src, bf16_t* __restrict__ dst) {
  const size_t i = ((size_t)blockIdx.x * 256 + threadIdx.x) * 8;
  v4f g0 = *(const v4f*)(src + i);
  v4f g1 = *(const v4f*)(src + i + 4);
  v8bf o;
#pragma unroll
  for (int j = 0; j < 4; ++j) {
    o[j]     = (bf16_t)g0[j];
    o[4 + j] = (bf16_t)g1[j];
  }
  *(v8bf*)(dst + i) = o;
}

// ---------------- Kernel 1: QKV projections ----------------
// One wave computes a 64x64 output tile of Q, K, or V. C = X * W^T (bf16 in, f32 acc).
// Q,K written f32 (RoPE in f32 follows); V written bf16 transposed: Vt[b,h,d,s].
__global__ __launch_bounds__(256) void gemm_qkv_kernel(
    const bf16_t* __restrict__ Xb,
    const bf16_t* __restrict__ Wqb, const bf16_t* __restrict__ Wkb,
    const bf16_t* __restrict__ Wvb,
    float* __restrict__ Qf, float* __restrict__ Kf, bf16_t* __restrict__ Vt) {
  const int lane = threadIdx.x & 31;
  const int wv   = threadIdx.x >> 5;
  const int hi   = lane >> 4;
  const int lnm  = lane & 15;

  const int wid   = blockIdx.x * 8 + wv;   // 0..3071
  const int which = wid >> 10;             // 0=Q,1=K,2=V
  const int t     = wid & 1023;
  const int mbase = (t >> 4) * 64;
  const int nbase = (t & 15) * 64;
  const bf16_t* __restrict__ W = (which == 0) ? Wqb : ((which == 1) ? Wkb : Wvb);

  v8f acc[4][4] = {};
  for (int kk = 0; kk < D_MODEL; kk += 32) {
    v16bf a[4], bfr[4];
#pragma unroll
    for (int i = 0; i < 4; ++i)
      a[i] = load_a_frag_bf16(Xb + (size_t)(mbase + i * 16 + lnm) * D_MODEL + kk, hi);
#pragma unroll
    for (int j = 0; j < 4; ++j)
      bfr[j] = load_b_frag_bf16(W + (size_t)(nbase + j * 16 + lnm) * D_MODEL + kk, hi);
#pragma unroll
    for (int i = 0; i < 4; ++i)
#pragma unroll
      for (int j = 0; j < 4; ++j)
        acc[i][j] = wmma_bf16(a[i], bfr[j], acc[i][j]);
  }

#pragma unroll
  for (int i = 0; i < 4; ++i) {
#pragma unroll
    for (int j = 0; j < 4; ++j) {
#pragma unroll
      for (int r = 0; r < 8; ++r) {
        const int m = mbase + i * 16 + r + hi * 8;
        const int n = nbase + j * 16 + lnm;
        const float v = acc[i][j][r];
        if (which == 0) {
          Qf[(size_t)m * D_MODEL + n] = v;
        } else if (which == 1) {
          Kf[(size_t)m * D_MODEL + n] = v;
        } else {
          const int bb = m >> 11, s = m & (SEQ - 1);
          const int hh = n >> 6,  d = n & (DK - 1);
          Vt[(((size_t)(bb * NH + hh) * DK + d) << 11) + s] = (bf16_t)v;
        }
      }
    }
  }
}

// ---------------- Kernel 2: RoPE + f32->bf16 ----------------
__global__ __launch_bounds__(256) void rope_convert_kernel(
    const float* __restrict__ Qf, const float* __restrict__ Kf,
    const int* __restrict__ tpos,
    bf16_t* __restrict__ Qb, bf16_t* __restrict__ Kb) {
  const int idx = blockIdx.x * 256 + threadIdx.x;  // 2^21 pairs total
  const int p = idx & 31;
  const int h = (idx >> 5) & (NH - 1);
  const int s = (idx >> 9) & (SEQ - 1);
  const int b = idx >> 20;
  const size_t base = ((size_t)(b * SEQ + s)) * D_MODEL + h * DK + 2 * p;

  const float pos  = (float)tpos[s];
  // freq = 10000^(-2p/64) = exp2(-p/32 * log2(10000))
  const float freq = exp2f(-(float)p * (13.287712379549449f / 32.0f));
  const float ang  = pos * freq;
  const float c = cosf(ang), sn = sinf(ang);

  const float q1 = Qf[base], q2 = Qf[base + 1];
  Qb[base]     = (bf16_t)(q1 * c - q2 * sn);
  Qb[base + 1] = (bf16_t)(q1 * sn + q2 * c);
  const float k1 = Kf[base], k2 = Kf[base + 1];
  Kb[base]     = (bf16_t)(k1 * c - k2 * sn);
  Kb[base + 1] = (bf16_t)(k1 * sn + k2 * c);
}

// ---------------- Kernel 3: causal flash attention ----------------
// One wave per (batch, head, 16-query tile). 8 waves / block.
__global__ __launch_bounds__(256) void attn_kernel(
    const bf16_t* __restrict__ Qb, const bf16_t* __restrict__ Kb,
    const bf16_t* __restrict__ Vt, bf16_t* __restrict__ Ob) {
  __shared__ __align__(32) bf16_t pbuf[8][16 * 32];

  const int lane = threadIdx.x & 31;
  const int wv   = threadIdx.x >> 5;
  const int hi   = lane >> 4;
  const int lnm  = lane & 15;

  const int id = blockIdx.x * 8 + wv;       // 0..4095
  const int qt = id & 127;
  const int h  = (id >> 7) & (NH - 1);
  const int b  = id >> 11;

  // Q A-fragments (RoPE already applied), d_k=64 -> two 32-deep chunks
  const size_t rowQ = ((size_t)(b * SEQ + qt * 16 + lnm)) * D_MODEL + h * DK;
  const v16bf aq0 = load_a_frag_bf16(Qb + rowQ, hi);
  const v16bf aq1 = load_a_frag_bf16(Qb + rowQ + 32, hi);

  v8f o0 = {}, o1 = {}, o2 = {}, o3 = {};
  float rmax[8], rsum[8];
#pragma unroll
  for (int r = 0; r < 8; ++r) { rmax[r] = -1e30f; rsum[r] = 0.0f; }

  bf16_t* __restrict__ pb = &pbuf[wv][0];
  const int q_last = qt * 16 + 15;

  for (int kb = 0; kb <= q_last; kb += 32) {
    // ---- scores: S[16q x 32k] as two 16x16 f32 tiles ----
    v8f s0 = {}, s1 = {};
    const size_t rk0 = ((size_t)(b * SEQ + kb + lnm)) * D_MODEL + h * DK + (hi ? 16 : 0);
    const size_t rk1 = rk0 + (size_t)16 * D_MODEL;
    s0 = wmma_bf16(aq0, *(const v16bf*)(Kb + rk0), s0);
    s0 = wmma_bf16(aq1, *(const v16bf*)(Kb + rk0 + 32), s0);
    s1 = wmma_bf16(aq0, *(const v16bf*)(Kb + rk1), s1);
    s1 = wmma_bf16(aq1, *(const v16bf*)(Kb + rk1 + 32), s1);

    // ---- online softmax (scale 1/sqrt(64)=0.125, causal mask) ----
#pragma unroll
    for (int r = 0; r < 8; ++r) {
      const int qm = qt * 16 + r + hi * 8;
      const int k0 = kb + lnm, k1 = kb + 16 + lnm;
      const float v0 = (k0 <= qm) ? s0[r] * 0.125f : -1e30f;
      const float v1 = (k1 <= qm) ? s1[r] * 0.125f : -1e30f;
      float mn = halfmax16(fmaxf(v0, v1));
      mn = fmaxf(mn, rmax[r]);
      const float alpha = __expf(rmax[r] - mn);
      const float p0 = __expf(v0 - mn);
      const float p1 = __expf(v1 - mn);
      rsum[r] = rsum[r] * alpha + halfsum16(p0 + p1);
      rmax[r] = mn;
      o0[r] = o0[r] * alpha; o1[r] = o1[r] * alpha;
      o2[r] = o2[r] * alpha; o3[r] = o3[r] * alpha;
      // stage P (bf16) to this wave's LDS slice, row-major [16][32]
      pb[(r + hi * 8) * 32 + lnm]      = (bf16_t)p0;
      pb[(r + hi * 8) * 32 + 16 + lnm] = (bf16_t)p1;
    }
    asm volatile("s_wait_dscnt 0" ::: "memory");

    // ---- O += P * V ; Vt rows (dim-major) give contiguous B-fragments ----
    const v16bf ap = load_a_frag_bf16(pb + lnm * 32, hi);
    const bf16_t* __restrict__ vb =
        Vt + (((size_t)(b * NH + h) * DK + lnm) << 11) + kb + (hi ? 16 : 0);
    o0 = wmma_bf16(ap, *(const v16bf*)(vb), o0);
    o1 = wmma_bf16(ap, *(const v16bf*)(vb + ((size_t)16 << 11)), o1);
    o2 = wmma_bf16(ap, *(const v16bf*)(vb + ((size_t)32 << 11)), o2);
    o3 = wmma_bf16(ap, *(const v16bf*)(vb + ((size_t)48 << 11)), o3);
  }

  // ---- normalize and write attention output (bf16, [B,S,D] layout) ----
#pragma unroll
  for (int r = 0; r < 8; ++r) {
    const float inv = 1.0f / rsum[r];
    const size_t orow =
        ((size_t)(b * SEQ + qt * 16 + r + hi * 8)) * D_MODEL + h * DK + lnm;
    Ob[orow]      = (bf16_t)(o0[r] * inv);
    Ob[orow + 16] = (bf16_t)(o1[r] * inv);
    Ob[orow + 32] = (bf16_t)(o2[r] * inv);
    Ob[orow + 48] = (bf16_t)(o3[r] * inv);
  }
}

// ---------------- Kernel 4: output projection ----------------
__global__ __launch_bounds__(256) void gemm_out_kernel(
    const bf16_t* __restrict__ Ob, const bf16_t* __restrict__ Wob,
    float* __restrict__ out) {
  const int lane = threadIdx.x & 31;
  const int wv   = threadIdx.x >> 5;
  const int hi   = lane >> 4;
  const int lnm  = lane & 15;

  const int wid   = blockIdx.x * 8 + wv;   // 0..1023
  const int mbase = (wid >> 4) * 64;
  const int nbase = (wid & 15) * 64;

  v8f acc[4][4] = {};
  for (int kk = 0; kk < D_MODEL; kk += 32) {
    v16bf a[4], bfr[4];
#pragma unroll
    for (int i = 0; i < 4; ++i)
      a[i] = load_a_frag_bf16(Ob + (size_t)(mbase + i * 16 + lnm) * D_MODEL + kk, hi);
#pragma unroll
    for (int j = 0; j < 4; ++j)
      bfr[j] = load_b_frag_bf16(Wob + (size_t)(nbase + j * 16 + lnm) * D_MODEL + kk, hi);
#pragma unroll
    for (int i = 0; i < 4; ++i)
#pragma unroll
      for (int j = 0; j < 4; ++j)
        acc[i][j] = wmma_bf16(a[i], bfr[j], acc[i][j]);
  }

#pragma unroll
  for (int i = 0; i < 4; ++i)
#pragma unroll
    for (int j = 0; j < 4; ++j)
#pragma unroll
      for (int r = 0; r < 8; ++r) {
        const int m = mbase + i * 16 + r + hi * 8;
        const int n = nbase + j * 16 + lnm;
        out[(size_t)m * D_MODEL + n] = acc[i][j][r];
      }
}

// ---------------- launch ----------------
extern "C" void kernel_launch(void* const* d_in, const int* in_sizes, int n_in,
                              void* d_out, int out_size, void* d_ws, size_t ws_size,
                              hipStream_t stream) {
  const float* x   = (const float*)d_in[0];
  const int*   tp  = (const int*)d_in[1];
  const float* Wq  = (const float*)d_in[2];
  const float* Wk  = (const float*)d_in[3];
  const float* Wv  = (const float*)d_in[4];
  const float* Wo  = (const float*)d_in[5];

  char* ws = (char*)d_ws;
  const size_t XMATF  = (size_t)NTOK * D_MODEL * sizeof(float);     // 16 MiB
  const size_t XMATB  = (size_t)NTOK * D_MODEL * sizeof(bf16_t);    //  8 MiB
  const size_t WMATB  = (size_t)D_MODEL * D_MODEL * sizeof(bf16_t); //  2 MiB

  size_t off = 0;
  bf16_t* Xb  = (bf16_t*)(ws + off); off += XMATB;
  bf16_t* Wqb = (bf16_t*)(ws + off); off += WMATB;
  bf16_t* Wkb = (bf16_t*)(ws + off); off += WMATB;
  bf16_t* Wvb = (bf16_t*)(ws + off); off += WMATB;
  bf16_t* Wob = (bf16_t*)(ws + off); off += WMATB;
  float*  Qf  = (float*)(ws + off);  off += XMATF;
  float*  Kf  = (float*)(ws + off);  off += XMATF;
  bf16_t* Qb  = (bf16_t*)(ws + off); off += XMATB;
  bf16_t* Kb  = (bf16_t*)(ws + off); off += XMATB;
  bf16_t* Vt  = (bf16_t*)(ws + off); off += XMATB;
  bf16_t* Ob  = (bf16_t*)(ws + off); off += XMATB;

  // one-time precision conversion (removes f32 loads + cvt from GEMM hot loops)
  convert_f32_bf16_kernel<<<2048, 256, 0, stream>>>(x,  Xb);   // 4096*1024 elems
  convert_f32_bf16_kernel<<<512,  256, 0, stream>>>(Wq, Wqb);  // 1024*1024 elems
  convert_f32_bf16_kernel<<<512,  256, 0, stream>>>(Wk, Wkb);
  convert_f32_bf16_kernel<<<512,  256, 0, stream>>>(Wv, Wvb);
  convert_f32_bf16_kernel<<<512,  256, 0, stream>>>(Wo, Wob);

  // 3 weights * 64 mtiles * 16 ntiles = 3072 wave-tiles, 8 waves/block
  gemm_qkv_kernel<<<384, 256, 0, stream>>>(Xb, Wqb, Wkb, Wvb, Qf, Kf, Vt);
  // 2^21 rotation pairs
  rope_convert_kernel<<<8192, 256, 0, stream>>>(Qf, Kf, tp, Qb, Kb);
  // 2*16*128 = 4096 (b,h,qtile) waves, 8 waves/block
  attn_kernel<<<512, 256, 0, stream>>>(Qb, Kb, Vt, Ob);
  // 64*16 = 1024 wave-tiles
  gemm_out_kernel<<<128, 256, 0, stream>>>(Ob, Wob, (float*)d_out);
}
